// CLSTransformerAggregator_19061064860083
// MI455X (gfx1250) — matrix operand order repaired and checked by reading, not compiled, not run on hardware
//
#include <hip/hip_runtime.h>
#include <hip/hip_bf16.h>
#include <math.h>

#define B_   8
#define N_   1024
#define D_   256
#define L_   4
#define H_   8
#define DFF_ 1024
#define DH_  32
#define TOK_ (N_ + 1)      // 1025 tokens (CLS + N)
#define M_   (B_ * TOK_)   // 8200 rows in the token dimension

typedef __attribute__((ext_vector_type(16))) __bf16 v16bf;
typedef __attribute__((ext_vector_type(8)))  float  v8f;

union FragBF { unsigned int u[8]; __bf16 h[16]; v16bf v; };
union FragF  { float f[8]; v8f v; };

__device__ inline v8f wmma_bf16(v16bf a, v16bf b, v8f c) {
#if defined(__HIP_DEVICE_COMPILE__)
  return __builtin_amdgcn_wmma_f32_16x16x32_bf16(false, a, false, b, (short)0, c,
                                                 false, false);
#else
  return c;
#endif
}

// ---------------------------------------------------------------------------
// fp32 -> bf16 elementwise convert (weights re-converted every call: no caching)
// ---------------------------------------------------------------------------
__global__ void f2bf_kernel(const float* __restrict__ in, __bf16* __restrict__ out, int n) {
  int i = blockIdx.x * blockDim.x + threadIdx.x;
  int stride = gridDim.x * blockDim.x;
  for (; i < n; i += stride) out[i] = (__bf16)in[i];
}

// ---------------------------------------------------------------------------
// x[b,t,:] = (t==0 ? cls : features[b,t-1]) + pos[t]
// ---------------------------------------------------------------------------
__global__ void embed_kernel(const float* __restrict__ feat, const float* __restrict__ cls,
                             const float* __restrict__ pos, float* __restrict__ X) {
  int idx = blockIdx.x * blockDim.x + threadIdx.x;
  if (idx >= M_ * D_) return;
  int tok = idx / D_, d = idx - tok * D_;
  int b = tok / TOK_, t = tok - b * TOK_;
  float v = (t == 0) ? cls[d] : feat[((size_t)b * N_ + (t - 1)) * D_ + d];
  X[idx] = v + pos[(size_t)t * D_ + d];
}

// ---------------------------------------------------------------------------
// LayerNorm fp32 -> bf16, one wave32 per row of 256
// ---------------------------------------------------------------------------
__global__ __launch_bounds__(256)
void layernorm_bf16(const float* __restrict__ X, const float* __restrict__ g,
                    const float* __restrict__ bta, __bf16* __restrict__ Out, int Mrows) {
  int wave = (blockIdx.x * blockDim.x + threadIdx.x) >> 5;
  int lane = threadIdx.x & 31;
  if (wave >= Mrows) return;
  const float* xr = X + (size_t)wave * D_;
  float lv[8];
  float s = 0.f;
#pragma unroll
  for (int j = 0; j < 8; ++j) { lv[j] = xr[lane + 32 * j]; s += lv[j]; }
#pragma unroll
  for (int off = 16; off > 0; off >>= 1) s += __shfl_xor(s, off, 32);
  float mean = s * (1.f / 256.f);
  float vs = 0.f;
#pragma unroll
  for (int j = 0; j < 8; ++j) { float d = lv[j] - mean; vs += d * d; }
#pragma unroll
  for (int off = 16; off > 0; off >>= 1) vs += __shfl_xor(vs, off, 32);
  float inv = rsqrtf(vs * (1.f / 256.f) + 1e-5f);
  __bf16* orow = Out + (size_t)wave * D_;
#pragma unroll
  for (int j = 0; j < 8; ++j) {
    int c = lane + 32 * j;
    orow[c] = (__bf16)((lv[j] - mean) * inv * g[c] + bta[c]);
  }
}

// ---------------------------------------------------------------------------
// Final LN on the 8 CLS rows -> fp32 d_out[0..B*D)
// ---------------------------------------------------------------------------
__global__ __launch_bounds__(256)
void final_ln_kernel(const float* __restrict__ X, const float* __restrict__ g,
                     const float* __restrict__ bta, float* __restrict__ out) {
  int w = threadIdx.x >> 5;      // batch index
  int lane = threadIdx.x & 31;
  const float* xr = X + (size_t)(w * TOK_) * D_;
  float lv[8];
  float s = 0.f;
#pragma unroll
  for (int j = 0; j < 8; ++j) { lv[j] = xr[lane + 32 * j]; s += lv[j]; }
#pragma unroll
  for (int off = 16; off > 0; off >>= 1) s += __shfl_xor(s, off, 32);
  float mean = s * (1.f / 256.f);
  float vs = 0.f;
#pragma unroll
  for (int j = 0; j < 8; ++j) { float d = lv[j] - mean; vs += d * d; }
#pragma unroll
  for (int off = 16; off > 0; off >>= 1) vs += __shfl_xor(vs, off, 32);
  float inv = rsqrtf(vs * (1.f / 256.f) + 1e-5f);
#pragma unroll
  for (int j = 0; j < 8; ++j) {
    int c = lane + 32 * j;
    out[w * D_ + c] = (lv[j] - mean) * inv * g[c] + bta[c];
  }
}

// ---------------------------------------------------------------------------
// WMMA GEMM:  C[m,n] = sum_k A[m,k] * W[n,k] + bias[n]
//   A: bf16 [Mrows, K] (lda), W: bf16 [Nout, K] (ldb, i.e. computes A @ W^T)
//   mode 0: store bf16      mode 1: store bf16 after exact GELU
//   mode 2: residual add fp32 into Xres
// One wave per 16x64 output tile: A fragment reused across 4 WMMAs per K-step
// (4x arithmetic intensity vs a 16x16 tile). K-loop step 32.
// ---------------------------------------------------------------------------
__global__ __launch_bounds__(32)
void gemm_wmma(const __bf16* __restrict__ A, int lda,
               const __bf16* __restrict__ W, int ldb,
               const float* __restrict__ bias,
               int Mrows, int K, int mode,
               __bf16* __restrict__ Obf, float* __restrict__ Xres, int ldo) {
  const int lane = threadIdx.x & 31;
  const int nn = lane & 15;
  const int half = lane >> 4;
  const int n0 = blockIdx.x * 64;     // 4 adjacent 16-wide N tiles
  const int m0 = blockIdx.y * 16;

  int arow = m0 + nn;
  if (arow > Mrows - 1) arow = Mrows - 1;   // clamp: garbage rows never stored
  const __bf16* Abase = A + (size_t)arow * lda;

  const __bf16* Wb[4];
#pragma unroll
  for (int t = 0; t < 4; ++t)
    Wb[t] = W + (size_t)(n0 + t * 16 + nn) * ldb + 16 * half;

  FragF acc[4];
#pragma unroll
  for (int t = 0; t < 4; ++t)
#pragma unroll
    for (int r = 0; r < 8; ++r) acc[t].f[r] = 0.f;

  for (int k0 = 0; k0 < K; k0 += 32) {
    FragBF a;
    const unsigned int* a32 = (const unsigned int*)(Abase + k0);
#pragma unroll
    for (int v = 0; v < 4; ++v) a.u[v] = a32[v + 8 * half];        // K = 0..7 (+16*half)
#pragma unroll
    for (int v = 4; v < 8; ++v) a.u[v] = a32[v + 8 * half];    // K = 8..15 (+16*half)

    FragBF b[4];
#pragma unroll
    for (int t = 0; t < 4; ++t) {
      const unsigned int* b32 = (const unsigned int*)(Wb[t] + k0);
#pragma unroll
      for (int v = 0; v < 8; ++v) b[t].u[v] = b32[v];              // K = 16*half .. +16
    }
#pragma unroll
    for (int t = 0; t < 4; ++t)
      acc[t].v = wmma_bf16(a.v, b[t].v, acc[t].v);
  }

#pragma unroll
  for (int t = 0; t < 4; ++t) {
    const float bv = bias[n0 + t * 16 + nn];
#pragma unroll
    for (int r = 0; r < 8; ++r) {
      int row = m0 + r + 8 * half;
      if (row >= Mrows) continue;
      float val = acc[t].f[r] + bv;
      if (mode == 1) val = 0.5f * val * (1.0f + erff(val * 0.70710678118654752f));
      if (mode == 2) {
        float* p = Xres + (size_t)row * ldo + n0 + t * 16 + nn;
        *p = *p + val;
      } else {
        Obf[(size_t)row * ldo + n0 + t * 16 + nn] = (__bf16)val;
      }
    }
  }
}

// ---------------------------------------------------------------------------
// Flash attention: one wave per (batch, head, 16-query tile).
// QK^T : one v_wmma_f32_16x16x32_bf16 per 16-key column tile (DH == 32 == K).
// Online softmax staged through a 16x32 LDS score tile; P@V via two bf16 WMMAs
// over each 32-key block.
// ---------------------------------------------------------------------------
__global__ __launch_bounds__(32)
void flash_attn(const __bf16* __restrict__ qkv, const int* __restrict__ mask,
                __bf16* __restrict__ Obf, float scale) {
  const int qt = blockIdx.x, h = blockIdx.y, b = blockIdx.z;
  const int lane = threadIdx.x & 31;
  const int nn = lane & 15;
  const int half = lane >> 4;

  __shared__ float s_lds[16][36];
  __shared__ float alpha_lds[16];
  __shared__ float l_lds[16];

  // ---- Q fragment (A-layout 16x32 bf16) ----
  int qrow = qt * 16 + nn;
  if (qrow > TOK_ - 1) qrow = TOK_ - 1;
  const unsigned int* q32 =
      (const unsigned int*)(qkv + ((size_t)(b * TOK_ + qrow) * 768 + h * 32));
  FragBF qa;
#pragma unroll
  for (int v = 0; v < 4; ++v) qa.u[v] = q32[v + 8 * half];
#pragma unroll
  for (int v = 4; v < 8; ++v) qa.u[v] = q32[v + 8 * half];

  FragF o0, o1;
#pragma unroll
  for (int r = 0; r < 8; ++r) { o0.f[r] = 0.f; o1.f[r] = 0.f; }
  float m_run = -1e30f, l_run = 0.f;   // valid in lanes 0..15 (row = lane)

  for (int kb = 0; kb < TOK_; kb += 32) {
    // ---- scores for 32 keys: two 16-key WMMAs ----
#pragma unroll
    for (int part = 0; part < 2; ++part) {
      int key = kb + part * 16 + nn;
      int kc = (key < TOK_) ? key : 0;
      const unsigned int* k32 = (const unsigned int*)(qkv +
          ((size_t)(b * TOK_ + kc) * 768 + 256 + h * 32 + 16 * half));
      FragBF kf;
#pragma unroll
      for (int v = 0; v < 8; ++v) kf.u[v] = k32[v];
      FragF sc;
#pragma unroll
      for (int r = 0; r < 8; ++r) sc.f[r] = 0.f;
      sc.v = wmma_bf16(qa.v, kf.v, sc.v);
      bool valid = (key < TOK_) && (key == 0 || mask[b * N_ + key - 1] != 0);
#pragma unroll
      for (int r = 0; r < 8; ++r)
        s_lds[r + 8 * half][part * 16 + nn] = valid ? sc.f[r] * scale : -1e30f;
    }
    __syncthreads();

    // ---- online softmax stats (lane m handles query row m) ----
    if (lane < 16) {
      float rmax = -1e30f;
#pragma unroll
      for (int j = 0; j < 32; ++j) rmax = fmaxf(rmax, s_lds[lane][j]);
      float mnew = fmaxf(m_run, rmax);
      float al = __expf(m_run - mnew);
      float rs = 0.f;
#pragma unroll
      for (int j = 0; j < 32; ++j) {
        float p = __expf(s_lds[lane][j] - mnew);
        s_lds[lane][j] = p;
        rs += p;
      }
      l_run = l_run * al + rs;
      m_run = mnew;
      alpha_lds[lane] = al;
    }
    __syncthreads();

    // ---- rescale O accumulators ----
#pragma unroll
    for (int r = 0; r < 8; ++r) {
      float f = alpha_lds[r + 8 * half];
      o0.f[r] *= f; o1.f[r] *= f;
    }

    // ---- P fragment (A-layout 16x32, read from LDS, cast to bf16) ----
    FragBF pa;
#pragma unroll
    for (int e = 0; e < 16; ++e) {
      int kk = e + 16 * half;
      pa.h[e] = (__bf16)s_lds[nn][kk];
    }

    // ---- V fragments (B-layout 32x16): dh cols 0..15 and 16..31 ----
    FragBF vb0, vb1;
#pragma unroll
    for (int e = 0; e < 16; ++e) {
      int kidx = kb + e + 16 * half;
      bool kv = kidx < TOK_;
      const __bf16* vp = qkv +
          ((size_t)(b * TOK_ + (kv ? kidx : 0)) * 768 + 512 + h * 32);
      vb0.h[e] = kv ? vp[nn]      : (__bf16)0.0f;
      vb1.h[e] = kv ? vp[nn + 16] : (__bf16)0.0f;
    }
    o0.v = wmma_bf16(pa.v, vb0.v, o0.v);
    o1.v = wmma_bf16(pa.v, vb1.v, o1.v);
    __syncthreads();   // protect s_lds for next iteration
  }

  if (lane < 16) l_lds[lane] = l_run;
  __syncthreads();
#pragma unroll
  for (int r = 0; r < 8; ++r) {
    int row = qt * 16 + r + 8 * half;
    if (row < TOK_) {
      float inv = 1.f / l_lds[r + 8 * half];
      size_t off = (size_t)(b * TOK_ + row) * D_ + h * 32;
      Obf[off + nn]      = (__bf16)(o0.f[r] * inv);
      Obf[off + nn + 16] = (__bf16)(o1.f[r] * inv);
    }
  }
}

// ---------------------------------------------------------------------------
// Last-layer CLS attention: recompute softmax(q0 . k) over 1025 keys per head,
// mean over heads, mask, renormalize. One block per batch.
// ---------------------------------------------------------------------------
__global__ __launch_bounds__(256)
void cls_attn_kernel(const __bf16* __restrict__ qkv, const int* __restrict__ mask,
                     float* __restrict__ out, float scale) {
  int b = blockIdx.x;
  int t = threadIdx.x;
  __shared__ float q0s[D_];
  __shared__ float sc[H_][1056];
  __shared__ float Mh[H_], Lh[H_];
  __shared__ float pbuf[N_];
  __shared__ float red[256];

  q0s[t] = (float)qkv[(size_t)(b * TOK_) * 768 + t];
  __syncthreads();

  for (int j = t; j < TOK_; j += 256) {
    bool valid = (j == 0) || (mask[b * N_ + j - 1] != 0);
    size_t kbase = (size_t)(b * TOK_ + j) * 768 + 256;
#pragma unroll
    for (int hh = 0; hh < H_; ++hh) {
      float dt = 0.f;
#pragma unroll
      for (int d = 0; d < DH_; ++d)
        dt += q0s[hh * DH_ + d] * (float)qkv[kbase + hh * DH_ + d];
      sc[hh][j] = valid ? dt * scale : -1e30f;
    }
  }
  __syncthreads();

  int w = t >> 5, lane = t & 31;   // wave w handles head w
  {
    float mx = -1e30f;
    for (int j = lane; j < TOK_; j += 32) mx = fmaxf(mx, sc[w][j]);
#pragma unroll
    for (int off = 16; off > 0; off >>= 1) mx = fmaxf(mx, __shfl_xor(mx, off, 32));
    float sm = 0.f;
    for (int j = lane; j < TOK_; j += 32) sm += __expf(sc[w][j] - mx);
#pragma unroll
    for (int off = 16; off > 0; off >>= 1) sm += __shfl_xor(sm, off, 32);
    if (lane == 0) { Mh[w] = mx; Lh[w] = sm; }
  }
  __syncthreads();

  float part = 0.f;
  for (int j = 1 + t; j <= N_; j += 256) {
    float v = 0.f;
#pragma unroll
    for (int hh = 0; hh < H_; ++hh) v += __expf(sc[hh][j] - Mh[hh]) / Lh[hh];
    v *= 0.125f;
    v *= (float)mask[b * N_ + j - 1];
    pbuf[j - 1] = v;
    part += v;
  }
  red[t] = part;
  __syncthreads();
  for (int s = 128; s > 0; s >>= 1) {
    if (t < s) red[t] += red[t + s];
    __syncthreads();
  }
  float inv = 1.f / (red[0] + 1e-8f);
  for (int j = 1 + t; j <= N_; j += 256) out[b * N_ + j - 1] = pbuf[j - 1] * inv;
}

// ---------------------------------------------------------------------------
extern "C" void kernel_launch(void* const* d_in, const int* in_sizes, int n_in,
                              void* d_out, int out_size, void* d_ws, size_t ws_size,
                              hipStream_t stream) {
  (void)in_sizes; (void)n_in; (void)out_size; (void)ws_size;
  const float* features = (const float*)d_in[0];
  const int*   mask     = (const int*)  d_in[1];
  const float* cls      = (const float*)d_in[2];
  const float* pos      = (const float*)d_in[3];
  const float* in_w     = (const float*)d_in[4];
  const float* in_b     = (const float*)d_in[5];
  const float* out_w    = (const float*)d_in[6];
  const float* out_b    = (const float*)d_in[7];
  const float* w1       = (const float*)d_in[8];
  const float* b1       = (const float*)d_in[9];
  const float* w2       = (const float*)d_in[10];
  const float* b2       = (const float*)d_in[11];
  const float* ln1g     = (const float*)d_in[12];
  const float* ln1b     = (const float*)d_in[13];
  const float* ln2g     = (const float*)d_in[14];
  const float* ln2b     = (const float*)d_in[15];
  const float* lnfg     = (const float*)d_in[16];
  const float* lnfb     = (const float*)d_in[17];

  char* ws = (char*)d_ws;
  float*  X    = (float*)ws;   ws += sizeof(float) * (size_t)M_ * D_;
  __bf16* XN   = (__bf16*)ws;  ws += 2ull * M_ * D_;
  __bf16* QKV  = (__bf16*)ws;  ws += 2ull * M_ * 3 * D_;
  __bf16* OATT = (__bf16*)ws;  ws += 2ull * M_ * D_;
  __bf16* HBUF = (__bf16*)ws;  ws += 2ull * M_ * DFF_;
  __bf16* WIN  = (__bf16*)ws;  ws += 2ull * L_ * 3 * D_ * D_;
  __bf16* WOUT = (__bf16*)ws;  ws += 2ull * L_ * D_ * D_;
  __bf16* W1B  = (__bf16*)ws;  ws += 2ull * L_ * DFF_ * D_;
  __bf16* W2B  = (__bf16*)ws;  ws += 2ull * L_ * D_ * DFF_;

  // weight conversion fp32 -> bf16 (every call; no cached state)
  f2bf_kernel<<<1024, 256, 0, stream>>>(in_w,  WIN,  L_ * 3 * D_ * D_);
  f2bf_kernel<<<1024, 256, 0, stream>>>(out_w, WOUT, L_ * D_ * D_);
  f2bf_kernel<<<1024, 256, 0, stream>>>(w1,    W1B,  L_ * DFF_ * D_);
  f2bf_kernel<<<1024, 256, 0, stream>>>(w2,    W2B,  L_ * D_ * DFF_);

  embed_kernel<<<(M_ * D_ + 255) / 256, 256, 0, stream>>>(features, cls, pos, X);

  const float scale = 0.17677669529663687f;   // 1/sqrt(DH)
  const int MT = (M_ + 15) / 16;              // 513 row tiles
  const int LNBLK = (M_ + 7) / 8;

  for (int l = 0; l < L_; ++l) {
    layernorm_bf16<<<LNBLK, 256, 0, stream>>>(X, ln1g + l * D_, ln1b + l * D_, XN, M_);
    gemm_wmma<<<dim3(3 * D_ / 64, MT), 32, 0, stream>>>(
        XN, D_, WIN + (size_t)l * 3 * D_ * D_, D_, in_b + l * 3 * D_,
        M_, D_, 0, QKV, nullptr, 3 * D_);
    flash_attn<<<dim3((TOK_ + 15) / 16, H_, B_), 32, 0, stream>>>(QKV, mask, OATT, scale);
    if (l == L_ - 1)
      cls_attn_kernel<<<B_, 256, 0, stream>>>(QKV, mask, (float*)d_out + B_ * D_, scale);
    gemm_wmma<<<dim3(D_ / 64, MT), 32, 0, stream>>>(
        OATT, D_, WOUT + (size_t)l * D_ * D_, D_, out_b + l * D_,
        M_, D_, 2, nullptr, X, D_);
    layernorm_bf16<<<LNBLK, 256, 0, stream>>>(X, ln2g + l * D_, ln2b + l * D_, XN, M_);
    gemm_wmma<<<dim3(DFF_ / 64, MT), 32, 0, stream>>>(
        XN, D_, W1B + (size_t)l * DFF_ * D_, D_, b1 + l * DFF_,
        M_, D_, 1, HBUF, nullptr, DFF_);
    gemm_wmma<<<dim3(D_ / 64, MT), 32, 0, stream>>>(
        HBUF, DFF_, W2B + (size_t)l * D_ * DFF_, DFF_, b2 + l * D_,
        M_, DFF_, 2, nullptr, X, D_);
  }

  final_ln_kernel<<<1, 256, 0, stream>>>(X, lnfg, lnfb, (float*)d_out);
}
